// MultiHeadedAttention_23897198035156
// MI455X (gfx1250) — matrix operand order repaired and checked by reading, not compiled, not run on hardware
//
#include <hip/hip_runtime.h>

typedef __attribute__((ext_vector_type(16))) __bf16 v16bf;
typedef __attribute__((ext_vector_type(8)))  float  v8f;

#define Hh 8
#define Dd 512
#define Ww 2048

__device__ __forceinline__ unsigned short f2bf(float f) {
  union { __bf16 h; unsigned short u; } c;
  c.h = (__bf16)f;                  // native v_cvt to bf16 (RTNE)
  return c.u;
}
__device__ __forceinline__ unsigned short bfval(float f) { return f2bf(f); }
__device__ __forceinline__ unsigned short bfval(unsigned short u) { return u; }

union Frag16 { v16bf v; uint4 q[2]; unsigned short u[16]; };
union Frag32 { v8f  v; float f[8]; };
union Pack16 { unsigned short u[16]; uint4 q[2]; };

__device__ __forceinline__ void storeC(float* p, float v) { *p = v; }
__device__ __forceinline__ void storeC(unsigned short* p, float v) { *p = f2bf(v); }

// Generic bf16-WMMA GEMM: C[M,N] = scale * A[M,K] x B[K,N] (+ bias[m])
// A,B fp32 (converted to bf16 while staging) or bf16 (ushort).
// TRB: logical B[k][n] is read from global Bg[n][k] (row-major N x K).
// TRC: store C[n*ldc + m] (bf16, packed 16B) instead of C[m*ldc + n].
// Per-slice base offset = (s>>3)*x_bs + (s&7)*x_hs with s = blockIdx.z.
template<typename TA, typename TB, bool TRB, typename TC, bool TRC, bool BIAS>
__global__ __launch_bounds__(256)
void gemm_wmma(const TA* __restrict__ A, long long a_bs, long long a_hs, int lda,
               const TB* __restrict__ B, long long b_bs, long long b_hs, int ldb,
               TC* __restrict__ C,       long long c_bs, long long c_hs, int ldc,
               const float* __restrict__ bias, float scale, int K)
{
  constexpr int PAD = 40;                       // halves per LDS row (80B pitch)
  __shared__ unsigned short As[128 * PAD];      // [m][k] row-major, 16B-aligned chunks
  __shared__ unsigned short Bs[128 * PAD];      // [n][k] (!) so B frags are contiguous

  const int s = blockIdx.z;
  const TA* Ag = A + (long long)(s >> 3) * a_bs + (long long)(s & 7) * a_hs;
  const TB* Bg = B + (long long)(s >> 3) * b_bs + (long long)(s & 7) * b_hs;
  TC*       Cg = C + (long long)(s >> 3) * c_bs + (long long)(s & 7) * c_hs;

  const int tid  = threadIdx.x;
  const int lane = tid & 31;
  const int wave = tid >> 5;
  const int wm0  = (wave >> 2) * 64;   // 2 waves along M
  const int wn0  = (wave & 3) * 32;    // 4 waves along N
  const int bM   = blockIdx.y * 128;
  const int bN   = blockIdx.x * 128;
  const int g    = lane >> 4;
  const int lc   = lane & 15;

  // staging coordinates: thread t -> row (tid>>1), 16-wide k-chunk (tid&1)
  const int sr = tid >> 1;
  const int sc = (tid & 1) * 16;

  Frag32 acc[4][2];
  #pragma unroll
  for (int sm = 0; sm < 4; ++sm)
    #pragma unroll
    for (int sn = 0; sn < 2; ++sn)
      #pragma unroll
      for (int i = 0; i < 8; ++i) acc[sm][sn].f[i] = 0.0f;

  TA areg[16];
  TB breg[16];

  auto loadA = [&](int k0) {
    const TA* src = Ag + (long long)(bM + sr) * lda + k0 + sc;
    #pragma unroll
    for (int i = 0; i < 16; ++i) areg[i] = src[i];
  };
  auto loadB = [&](int k0) {
    if constexpr (TRB) {
      // Bg row-major N x K: contiguous along k
      const TB* src = Bg + (long long)(bN + sr) * ldb + k0 + sc;
      #pragma unroll
      for (int i = 0; i < 16; ++i) breg[i] = src[i];
    } else {
      // Bg row-major K x N: per-thread strided walk down column (coalesced across threads)
      const TB* src = Bg + (long long)(k0 + sc) * ldb + bN + sr;
      #pragma unroll
      for (int i = 0; i < 16; ++i) breg[i] = src[(long long)i * ldb];
    }
  };
  auto stage = [&]() {
    Pack16 ta, tb;
    #pragma unroll
    for (int i = 0; i < 16; ++i) { ta.u[i] = bfval(areg[i]); tb.u[i] = bfval(breg[i]); }
    uint4* ad = (uint4*)&As[sr * PAD + sc];
    ad[0] = ta.q[0]; ad[1] = ta.q[1];
    uint4* bd = (uint4*)&Bs[sr * PAD + sc];
    bd[0] = tb.q[0]; bd[1] = tb.q[1];
  };

  const int ksteps = K >> 5;
  loadA(0); loadB(0);

  for (int t = 0; t < ksteps; ++t) {
    stage();
    __syncthreads();
    if (t + 1 < ksteps) { loadA((t + 1) * 32); loadB((t + 1) * 32); }

    // ---- fragments (ISA 7.12.2): all contiguous 16B LDS reads ----
    Frag16 afr[4];
    #pragma unroll
    for (int sm = 0; sm < 4; ++sm) {
      const unsigned short* row = &As[(wm0 + sm * 16 + lc) * PAD];
      afr[sm].q[0] = *(const uint4*)(row + g * 8);        // k = g*8 .. g*8+7
      afr[sm].q[1] = *(const uint4*)(row + 16 + g * 8);   // k = 16+g*8 ..
    }
    Frag16 bfr[2];
    #pragma unroll
    for (int sn = 0; sn < 2; ++sn) {
      const unsigned short* row = &Bs[(wn0 + sn * 16 + lc) * PAD];
      bfr[sn].q[0] = *(const uint4*)(row + g * 8);        // k = g*8 .. g*8+7
      bfr[sn].q[1] = *(const uint4*)(row + 16 + g * 8);   // k = 16+g*8 ..
    }

    #pragma unroll
    for (int sm = 0; sm < 4; ++sm)
      #pragma unroll
      for (int sn = 0; sn < 2; ++sn)
        acc[sm][sn].v = __builtin_amdgcn_wmma_f32_16x16x32_bf16(
            false, afr[sm].v, false, bfr[sn].v,
            (short)0, acc[sm][sn].v, false, false);
    __syncthreads();
  }

  // ---- epilogue: C frag layout (lane 0-15: M=r, lane 16-31: M=8+r; N=lane&15) ----
  #pragma unroll
  for (int sm = 0; sm < 4; ++sm)
    #pragma unroll
    for (int sn = 0; sn < 2; ++sn) {
      const int mb = bM + wm0 + sm * 16 + g * 8;
      const int n  = bN + wn0 + sn * 16 + lc;
      if constexpr (TRC) {
        // packed 8 x bf16 = 16B contiguous along m
        Pack16 t;
        #pragma unroll
        for (int r = 0; r < 8; ++r) {
          float val = acc[sm][sn].f[r] * scale;
          if constexpr (BIAS) val += bias[mb + r];
          t.u[r] = f2bf(val);
        }
        *(uint4*)&Cg[(long long)n * ldc + mb] = t.q[0];
      } else {
        #pragma unroll
        for (int r = 0; r < 8; ++r) {
          float val = acc[sm][sn].f[r] * scale;
          if constexpr (BIAS) val += bias[mb + r];
          storeC(&Cg[(long long)(mb + r) * ldc + n], val);
        }
      }
    }
}

// Row softmax: one 256-thread block per row of 2048 fp32, in place.
__global__ __launch_bounds__(256)
void softmax_rows(float* __restrict__ S)
{
  __shared__ float red[256];
  const int t = threadIdx.x;
  float* p = S + (long long)blockIdx.x * 2048;

  float v[8];
  float mx = -1e30f;
  #pragma unroll
  for (int i = 0; i < 8; ++i) { v[i] = p[t + i * 256]; mx = fmaxf(mx, v[i]); }
  red[t] = mx; __syncthreads();
  for (int o = 128; o > 0; o >>= 1) { if (t < o) red[t] = fmaxf(red[t], red[t + o]); __syncthreads(); }
  mx = red[0]; __syncthreads();

  float sum = 0.0f;
  #pragma unroll
  for (int i = 0; i < 8; ++i) { v[i] = __expf(v[i] - mx); sum += v[i]; }
  red[t] = sum; __syncthreads();
  for (int o = 128; o > 0; o >>= 1) { if (t < o) red[t] += red[t + o]; __syncthreads(); }
  const float inv = 1.0f / red[0];

  #pragma unroll
  for (int i = 0; i < 8; ++i) p[t + i * 256] = v[i] * inv;
}

extern "C" void kernel_launch(void* const* d_in, const int* in_sizes, int n_in,
                              void* d_out, int out_size, void* d_ws, size_t ws_size,
                              hipStream_t stream)
{
  (void)in_sizes; (void)n_in; (void)out_size; (void)ws_size;

  const float* q  = (const float*)d_in[0];
  const float* kx = (const float*)d_in[1];
  const float* vx = (const float*)d_in[2];
  const float* wq = (const float*)d_in[3];
  const float* bq = (const float*)d_in[4];
  const float* wk = (const float*)d_in[5];
  const float* bk = (const float*)d_in[6];
  const float* wv = (const float*)d_in[7];
  const float* bv = (const float*)d_in[8];
  const float* wm = (const float*)d_in[9];
  const float* bm = (const float*)d_in[10];
  float* out = (float*)d_out;

  const size_t SL = (size_t)Dd * Ww;           // 1048576 elems per slice
  unsigned short* Qt = (unsigned short*)d_ws;  // [16][2048][512] bf16 (transposed)
  unsigned short* Kb = Qt + 16 * SL;           // [16][512][2048] bf16
  unsigned short* Vb = Kb + 16 * SL;           // [16][512][2048] bf16
  unsigned short* Xb = Vb + 16 * SL;           // [16][512][2048] bf16
  float*          Sc = (float*)(Xb + 16 * SL); // [2048][2048] fp32 (per-slice reuse)

  const long long IB  = (long long)Dd * Hh * Ww;  // input/output batch stride
  const long long IH  = Ww;                       // input/output h stride
  const int       ILD = Hh * Ww;                  // channel stride = 16384
  const long long WSb = 8LL * (long long)SL;
  const long long WSh = (long long)SL;
  const float inv_sqrt_d = 0.04419417382415922f;  // 1/sqrt(512)

  dim3 blk(256);
  dim3 gProj(16, 4, 16);    // N=2048/128, M=512/128, 16 slices

  // Q projection -> Qt stored transposed [n][d]
  gemm_wmma<float, float, false, unsigned short, true, true><<<gProj, blk, 0, stream>>>(
      wq, 0LL, 0LL, 512,  q, IB, IH, ILD,  Qt, WSb, WSh, 512,  bq, 1.0f, 512);
  // K projection -> Kb [d][m]
  gemm_wmma<float, float, false, unsigned short, false, true><<<gProj, blk, 0, stream>>>(
      wk, 0LL, 0LL, 512,  kx, IB, IH, ILD,  Kb, WSb, WSh, 2048,  bk, 1.0f, 512);
  // V projection -> Vb [d][m]
  gemm_wmma<float, float, false, unsigned short, false, true><<<gProj, blk, 0, stream>>>(
      wv, 0LL, 0LL, 512,  vx, IB, IH, ILD,  Vb, WSb, WSh, 2048,  bv, 1.0f, 512);

  for (int s = 0; s < 16; ++s) {
    // scores[n][m] = (Qt_s x Kb_s) / sqrt(512) ; M=N=2048, K=512
    dim3 gS(16, 16, 1);
    gemm_wmma<unsigned short, unsigned short, false, float, false, false><<<gS, blk, 0, stream>>>(
        Qt + s * SL, 0LL, 0LL, 512,
        Kb + s * SL, 0LL, 0LL, 2048,
        Sc, 0LL, 0LL, 2048, nullptr, inv_sqrt_d, 512);

    softmax_rows<<<dim3(2048), blk, 0, stream>>>(Sc);

    // X_s[d][n] = Vb_s x P^T ; M=512, N=2048, K=2048; B read transposed from Sc
    dim3 gO(16, 4, 1);
    gemm_wmma<unsigned short, float, true, unsigned short, false, false><<<gO, blk, 0, stream>>>(
        Vb + s * SL, 0LL, 0LL, 2048,
        Sc, 0LL, 0LL, 2048,
        Xb + s * SL, 0LL, 0LL, 2048, nullptr, 1.0f, 2048);
  }

  // Final projection: out[b,e,h,w] = Wm x X + bm (fp32 output, original strides)
  gemm_wmma<float, unsigned short, false, float, false, true><<<gProj, blk, 0, stream>>>(
      wm, 0LL, 0LL, 512,  Xb, WSb, WSh, 2048,  out, IB, IH, ILD,  bm, 1.0f, 512);
}
